// PRWKVBlock_68805376082240
// MI455X (gfx1250) — matrix operand order repaired and checked by reading, not compiled, not run on hardware
//
#include <hip/hip_runtime.h>
#include <math.h>

// ---------------- problem constants ----------------
#define Bb    4
#define Nn    2048
#define Cc    384
#define Hh    8
#define Dd    48
#define KNNk  16
#define ITER  3
#define HIDc  1536
#define BN    (Bb*Nn)                 // 8192 tokens
#define BNC   ((size_t)BN*Cc)         // 3145728
#define BNK   (BN*KNNk)               // 131072 edge rows
#define EK    387                     // 3 + C
#define EKP   416                     // padded edge row stride (13*32)

typedef __attribute__((ext_vector_type(16))) __bf16 v16bf;
typedef __attribute__((ext_vector_type(8)))  float  v8f;

__device__ __forceinline__ float sigf(float x){ return 1.f/(1.f+expf(-x)); }
__device__ __forceinline__ float epi_f(float x, int e){
  if(e==1) return fmaxf(x,0.f);                              // relu
  if(e==2) return sigf(x);                                   // sigmoid
  if(e==3) return x*sigf(x);                                 // silu
  if(e==4) return 0.5f*x*(1.f+erff(x*0.7071067811865475f));  // exact gelu
  return x;
}

// ---------------------------------------------------------------------------
// Weight pre-pack into WMMA B-fragment layout (bf16), zero-padded:
// fragment (kt,nt) occupies 512 bf16 at dst + (kt*NTp+nt)*512; within it,
// lane L's 16 values live at +L*16:  lanes0-15 -> N=nt*16+L, K=kt*32+{0..7,16..23};
// lanes16-31 -> N=nt*16+L-16, K=kt*32+{8..15,24..31} (2 bf16 per dword).
// ---------------------------------------------------------------------------
__global__ void k_packB(const float* __restrict__ src, __bf16* __restrict__ dst,
                        int K, int N, int NTp){
  const int tile = blockIdx.x;              // kt*NTp + nt
  const int nt = tile % NTp;
  const int tid = threadIdx.x;
  const int lane = tid >> 3, j = tid & 7;
  const int kb = (tile / NTp)*32 + 8*(lane>>4) + (j >= 4 ? 16 + 2*(j-4) : 2*j);
  const int n  = nt*16 + (lane & 15);
  const float b0 = (kb   < K && n < N) ? src[(size_t)kb*N + n]     : 0.f;
  const float b1 = (kb+1 < K && n < N) ? src[(size_t)(kb+1)*N + n] : 0.f;
  const size_t o = ((size_t)tile*32 + lane)*16 + 2*j;
  dst[o] = (__bf16)b0; dst[o+1] = (__bf16)b1;
}

// ---------------------------------------------------------------------------
// WMMA GEMM: D[M,N] = epi( (acc? D : 0) + A' @ B + bias ), A' = A + coef[k]*XS
// M multiple of 128 (always true here). K bounds handled by zero-padded
// packed B (and zero-padded A rows for K=387). Wave computes a 16x64 strip:
// one A fragment feeds 4 v_wmma_f32_16x16x32_bf16. Block = 8 waves = 128x64.
// A fragment = 4 unconditional float4 loads (two 32B runs per lane).
// ---------------------------------------------------------------------------
template<int MIX>
__global__ __launch_bounds__(256) void k_gemm_t(
    const float* __restrict__ A, const float* __restrict__ XS,
    const float* __restrict__ coef, const __bf16* __restrict__ Bp,
    const float* __restrict__ bias, float* __restrict__ D,
    int N, int KT, int NTp, int lda, int ldd, int acc, int epi)
{
  const int wave = threadIdx.x >> 5, lane = threadIdx.x & 31;
  const int half = lane >> 4,        lm   = lane & 15;
  const int rowBase = blockIdx.y*128 + wave*16;
  const int ntBase  = blockIdx.x*4;

  const int arow = rowBase + lm;
  const float* Ar = A + (size_t)arow*lda;
  const float* Xr = MIX ? (XS + (size_t)arow*lda) : nullptr;

  v8f c0 = {0.f,0.f,0.f,0.f,0.f,0.f,0.f,0.f};
  v8f c1 = c0, c2 = c0, c3 = c0;

  for (int kt = 0; kt < KT; ++kt) {
    const int base = kt*32 + 8*half;
    __builtin_prefetch(Ar + kt*32 + 32, 0, 1);   // speculative, branch-free
    // ---- A fragment: lane holds K runs [base,base+8) and [base+16,base+24)
    const float4* Ap = (const float4*)(Ar + base);
    float4 p0 = Ap[0], p1 = Ap[1], q0 = Ap[4], q1 = Ap[5];
    if (MIX) {
      const float4* Xp = (const float4*)(Xr + base);
      const float4* Cp = (const float4*)(coef + base);
      const float4 x0 = Xp[0], x1 = Xp[1], x2 = Xp[4], x3 = Xp[5];
      const float4 g0 = Cp[0], g1 = Cp[1], g2 = Cp[4], g3 = Cp[5];
      p0.x += g0.x*x0.x; p0.y += g0.y*x0.y; p0.z += g0.z*x0.z; p0.w += g0.w*x0.w;
      p1.x += g1.x*x1.x; p1.y += g1.y*x1.y; p1.z += g1.z*x1.z; p1.w += g1.w*x1.w;
      q0.x += g2.x*x2.x; q0.y += g2.y*x2.y; q0.z += g2.z*x2.z; q0.w += g2.w*x2.w;
      q1.x += g3.x*x3.x; q1.y += g3.y*x3.y; q1.z += g3.z*x3.z; q1.w += g3.w*x3.w;
    }
    v16bf af;
    af[0]=(__bf16)p0.x;  af[1]=(__bf16)p0.y;  af[2]=(__bf16)p0.z;  af[3]=(__bf16)p0.w;
    af[4]=(__bf16)p1.x;  af[5]=(__bf16)p1.y;  af[6]=(__bf16)p1.z;  af[7]=(__bf16)p1.w;
    af[8]=(__bf16)q0.x;  af[9]=(__bf16)q0.y;  af[10]=(__bf16)q0.z; af[11]=(__bf16)q0.w;
    af[12]=(__bf16)q1.x; af[13]=(__bf16)q1.y; af[14]=(__bf16)q1.z; af[15]=(__bf16)q1.w;
    // ---- B fragments: one 32B vector load per sub-tile, no guards ----
    const __bf16* Bfr = Bp + ((size_t)(kt*NTp + ntBase)*32 + lane)*16;
    const v16bf b0 = *(const v16bf*)(Bfr);
    const v16bf b1 = *(const v16bf*)(Bfr + 512);
    const v16bf b2 = *(const v16bf*)(Bfr + 1024);
    const v16bf b3 = *(const v16bf*)(Bfr + 1536);
    c0 = __builtin_amdgcn_wmma_f32_16x16x32_bf16(false, af, false, b0, (short)0, c0, false, false);
    c1 = __builtin_amdgcn_wmma_f32_16x16x32_bf16(false, af, false, b1, (short)0, c1, false, false);
    c2 = __builtin_amdgcn_wmma_f32_16x16x32_bf16(false, af, false, b2, (short)0, c2, false, false);
    c3 = __builtin_amdgcn_wmma_f32_16x16x32_bf16(false, af, false, b3, (short)0, c3, false, false);
  }

  v8f accs[4] = {c0, c1, c2, c3};
  #pragma unroll
  for (int s = 0; s < 4; ++s) {
    const int ncol = (ntBase + s)*16 + lm;
    if (ncol < N) {
      const float bi = bias ? bias[ncol] : 0.f;
      #pragma unroll
      for (int i = 0; i < 8; ++i) {
        const int row = rowBase + i + 8*half;
        float x = accs[s][i] + bi;
        if (acc) x += D[(size_t)row*ldd + ncol];
        D[(size_t)row*ldd + ncol] = epi_f(x, epi);
      }
    }
  }
}

// ------------------------- LayerNorm: one wave per row ----------------------
__global__ __launch_bounds__(256) void k_layernorm(
    const float* __restrict__ X, const float* __restrict__ g,
    const float* __restrict__ b, float* __restrict__ Y, int rows)
{
  const int wave = threadIdx.x >> 5, lane = threadIdx.x & 31;
  const int row  = blockIdx.x*8 + wave;
  if (row >= rows) return;
  const float* x = X + (size_t)row*Cc;
  float v[12], s = 0.f, ss = 0.f;
  #pragma unroll
  for (int j = 0; j < 12; ++j) { v[j] = x[lane + 32*j]; s += v[j]; ss += v[j]*v[j]; }
  #pragma unroll
  for (int m = 16; m > 0; m >>= 1) { s += __shfl_xor(s, m, 32); ss += __shfl_xor(ss, m, 32); }
  const float mu = s*(1.f/Cc);
  const float var = ss*(1.f/Cc) - mu*mu;
  const float r = rsqrtf(var + 1e-5f);
  #pragma unroll
  for (int j = 0; j < 12; ++j) {
    const int cch = lane + 32*j;
    Y[(size_t)row*Cc + cch] = (v[j]-mu)*r*g[cch] + b[cch];
  }
}

// --------------- BQE x_star: channel-quarter token rolls (+1,-1,+2) ---------
__global__ void k_xstar(const float* __restrict__ X, float* __restrict__ XS){
  size_t i = (size_t)blockIdx.x*256 + threadIdx.x;
  if (i >= BNC) return;
  const int c = (int)(i % Cc); const size_t bn = i / Cc;
  const int n = (int)(bn % Nn); const size_t b = bn / Nn;
  const int q = c / (Cc/4);
  const int shift = (q==0) ? 0 : (q==1 ? 1 : (q==2 ? -1 : 2));
  const int ns = (n - shift + Nn) % Nn;
  XS[i] = X[(b*Nn + ns)*Cc + c];
}

// ---- coef[k] = 1 - sigmoid(mu[k]) for the six BQE mixers -------------------
__global__ void k_coef(const float* m0,const float* m1,const float* m2,
                       const float* m3,const float* m4,const float* m5,
                       float* __restrict__ cf){
  const int i = threadIdx.x;
  if (i < Cc) {
    cf[0*Cc+i]=1.f-sigf(m0[i]); cf[1*Cc+i]=1.f-sigf(m1[i]);
    cf[2*Cc+i]=1.f-sigf(m2[i]); cf[3*Cc+i]=1.f-sigf(m3[i]);
    cf[4*Cc+i]=1.f-sigf(m4[i]); cf[5*Cc+i]=1.f-sigf(m5[i]);
  }
}

// ---- decay: w = exp(-exp(base + tanh(c*A)*B)) (in place on c buffer) -------
__global__ void k_decay(float* __restrict__ cw, const float* __restrict__ dA,
                        const float* __restrict__ dB, const float* __restrict__ db){
  size_t i = (size_t)blockIdx.x*256 + threadIdx.x;
  if (i >= BNC) return;
  const int c = (int)(i % Cc);
  const float nu = db[c] + tanhf(cw[i]*dA[c])*dB[c];
  cw[i] = expf(-expf(nu));
}

// -------- WKV directional scan: block per (b,h), lane d owns state row d ----
__global__ __launch_bounds__(64) void k_wkv(
    const float* __restrict__ R, const float* __restrict__ K,
    const float* __restrict__ V, const float* __restrict__ W,
    const float* __restrict__ U, float* __restrict__ O, int dir)
{
  const int b = blockIdx.x, h = blockIdx.y, d = threadIdx.x;
  __shared__ float sk[Dd], sv[Dd];
  float st[Dd];
  #pragma unroll
  for (int e = 0; e < Dd; ++e) st[e] = 0.f;
  const float ud = (d < Dd) ? U[h*Dd + d] : 0.f;
  for (int t = 0; t < Nn; ++t) {
    const int tt = dir ? (Nn-1-t) : t;
    const size_t base = ((size_t)b*Nn + tt)*Cc + h*Dd;
    float kd=0.f, wd=0.f, rd=0.f;
    if (d < Dd) { kd = K[base+d]; wd = W[base+d]; rd = R[base+d];
                  sk[d] = kd; sv[d] = V[base+d]; }
    __syncthreads();
    if (d < Dd) {
      float sc = 0.f, vs = 0.f;
      #pragma unroll
      for (int e = 0; e < Dd; ++e) {
        const float ke = sk[e], ve = sv[e];
        sc += st[e]*ke; vs += ve;
        st[e] = wd*st[e] + kd*ve;     // sc uses pre-update state
      }
      O[base+d] = sigf(rd)*(sc + ud*kd*vs);
    }
    __syncthreads();
  }
}

// ---- GroupNorm over (T, Dgroup) per (b,h) on 0.5*(fwd+bwd), times gate -----
__global__ __launch_bounds__(256) void k_gn(
    const float* __restrict__ F, const float* __restrict__ Bk,
    const float* __restrict__ G, const float* __restrict__ gg,
    const float* __restrict__ gb, float* __restrict__ Y)
{
  const int b = blockIdx.x, h = blockIdx.y, tid = threadIdx.x;
  __shared__ float r1[256], r2[256];
  float s = 0.f, ss = 0.f;
  for (int idx = tid; idx < Nn*Dd; idx += 256) {
    const int t = idx / Dd, d = idx % Dd;
    const size_t p = ((size_t)b*Nn + t)*Cc + h*Dd + d;
    const float v = 0.5f*(F[p] + Bk[p]);
    s += v; ss += v*v;
  }
  r1[tid] = s; r2[tid] = ss; __syncthreads();
  for (int m = 128; m > 0; m >>= 1) {
    if (tid < m) { r1[tid] += r1[tid+m]; r2[tid] += r2[tid+m]; }
    __syncthreads();
  }
  const float inv = 1.f/((float)Nn*Dd);
  const float mu = r1[0]*inv, var = r2[0]*inv - mu*mu;
  const float rinv = rsqrtf(var + 1e-5f);
  for (int idx = tid; idx < Nn*Dd; idx += 256) {
    const int t = idx / Dd, d = idx % Dd, cch = h*Dd + d;
    const size_t p = ((size_t)b*Nn + t)*Cc + cch;
    const float v = 0.5f*(F[p] + Bk[p]);
    Y[p] = ((v-mu)*rinv*gg[cch] + gb[cch]) * G[p];
  }
}

// ---------------- brute-force KNN (16 smallest of 2048) ---------------------
__global__ __launch_bounds__(128) void k_knn(const float* __restrict__ xyz,
                                             int* __restrict__ idx){
  __shared__ float sx[Nn], sy[Nn], sz[Nn];
  const int b = blockIdx.y;
  const float* P = xyz + (size_t)b*Nn*3;
  for (int j = threadIdx.x; j < Nn; j += 128) {
    sx[j] = P[j*3]; sy[j] = P[j*3+1]; sz[j] = P[j*3+2];
  }
  __syncthreads();
  const int i = blockIdx.x*128 + threadIdx.x;
  const float px = sx[i], py = sy[i], pz = sz[i];
  float bd[KNNk]; int bi[KNNk];
  #pragma unroll
  for (int s = 0; s < KNNk; ++s) { bd[s] = 3.4e38f; bi[s] = 0; }
  float worst = 3.4e38f; int wsl = 0;
  for (int j = 0; j < Nn; ++j) {
    const float dx = sx[j]-px, dy = sy[j]-py, dz = sz[j]-pz;
    const float dsq = dx*dx + dy*dy + dz*dz;
    if (dsq < worst) {
      bd[wsl] = dsq; bi[wsl] = j;
      worst = bd[0]; wsl = 0;
      #pragma unroll
      for (int s = 1; s < KNNk; ++s) if (bd[s] > worst) { worst = bd[s]; wsl = s; }
    }
  }
  for (int s = 0; s < KNNk; ++s) idx[((size_t)b*Nn + i)*KNNk + s] = bi[s];
}

// ---- assemble padded edge rows: [rel(3) | f[neighbor](384) | zeros] --------
__global__ __launch_bounds__(256) void k_edge(
    const float* __restrict__ xyz, const float* __restrict__ delta,
    const float* __restrict__ F, const int* __restrict__ idx,
    float* __restrict__ E)
{
  const int bn = blockIdx.x, b = bn >> 11, n = bn & (Nn-1);
  const float* pc = xyz + ((size_t)b*Nn + n)*3;
  const float cx = pc[0] + delta[(size_t)bn*4+0];
  const float cy = pc[1] + delta[(size_t)bn*4+1];
  const float cz = pc[2] + delta[(size_t)bn*4+2];
  for (int kk = 0; kk < KNNk; ++kk) {
    const int j = idx[(size_t)bn*KNNk + kk];
    const size_t row = (size_t)bn*KNNk + kk;
    const float* pj = xyz + ((size_t)b*Nn + j)*3;
    const float* fj = F   + ((size_t)b*Nn + j)*Cc;
    const float r0 = pj[0]-cx, r1 = pj[1]-cy, r2 = pj[2]-cz;
    for (int c = threadIdx.x; c < EKP; c += 256) {
      float v;
      if (c < 3)        v = (c==0) ? r0 : ((c==1) ? r1 : r2);
      else if (c < EK)  v = fj[c-3];
      else              v = 0.f;
      E[row*EKP + c] = v;
    }
  }
}

// ---- max over the 16 neighbors ---------------------------------------------
__global__ void k_rowmax(const float* __restrict__ E3, float* __restrict__ agg){
  size_t i = (size_t)blockIdx.x*256 + threadIdx.x;
  if (i >= BNC) return;
  const size_t bn = i / Cc; const int c = (int)(i % Cc);
  float m = -3.4e38f;
  #pragma unroll
  for (int kk = 0; kk < KNNk; ++kk)
    m = fmaxf(m, E3[(bn*KNNk + kk)*(size_t)Cc + c]);
  agg[i] = m;
}

__global__ void k_add2(float* __restrict__ y, const float* __restrict__ a){
  size_t i = (size_t)blockIdx.x*256 + threadIdx.x;
  if (i < BNC) y[i] += a[i];
}

__global__ void k_final(const float* __restrict__ x, const float* __restrict__ r,
                        const float* __restrict__ v, float* __restrict__ out){
  size_t i = (size_t)blockIdx.x*256 + threadIdx.x;
  if (i < BNC) out[i] = x[i] + r[i]*v[i];
}

// ===========================================================================
extern "C" void kernel_launch(void* const* d_in, const int* in_sizes, int n_in,
                              void* d_out, int out_size, void* d_ws, size_t ws_size,
                              hipStream_t stream)
{
  (void)in_sizes; (void)n_in; (void)out_size; (void)ws_size;
  int a = 0;
  const float* xyz       = (const float*)d_in[a++];
  const float* features  = (const float*)d_in[a++];
  const float* norm1_g   = (const float*)d_in[a++];
  const float* norm1_b   = (const float*)d_in[a++];
  const float* norm2_g   = (const float*)d_in[a++];
  const float* norm2_b   = (const float*)d_in[a++];
  const float* mu_r      = (const float*)d_in[a++];
  const float* mu_k      = (const float*)d_in[a++];
  const float* mu_v      = (const float*)d_in[a++];
  const float* mu_w      = (const float*)d_in[a++];
  const float* Wr        = (const float*)d_in[a++];
  const float* Wk        = (const float*)d_in[a++];
  const float* Wv        = (const float*)d_in[a++];
  const float* Wg        = (const float*)d_in[a++];
  const float* Ww        = (const float*)d_in[a++];
  const float* Wo        = (const float*)d_in[a++];
  const float* decay_A   = (const float*)d_in[a++];
  const float* decay_B   = (const float*)d_in[a++];
  const float* decay_bse = (const float*)d_in[a++];
  const float* uvec      = (const float*)d_in[a++];
  const float* gn_g      = (const float*)d_in[a++];
  const float* gn_b      = (const float*)d_in[a++];
  const float* off_W1    = (const float*)d_in[a++];
  const float* off_b1    = (const float*)d_in[a++];
  const float* off_W2    = (const float*)d_in[a++];
  const float* off_b2    = (const float*)d_in[a++];
  const float* edge_W1   = (const float*)d_in[a++];
  const float* edge_b1   = (const float*)d_in[a++];
  const float* edge_W2   = (const float*)d_in[a++];
  const float* edge_b2   = (const float*)d_in[a++];
  const float* upd_W1    = (const float*)d_in[a++];
  const float* upd_b1    = (const float*)d_in[a++];
  const float* upd_W2    = (const float*)d_in[a++];
  const float* upd_b2    = (const float*)d_in[a++];
  const float* proj_W1   = (const float*)d_in[a++];
  const float* proj_b1   = (const float*)d_in[a++];
  const float* proj_W2   = (const float*)d_in[a++];
  const float* proj_b2   = (const float*)d_in[a++];
  const float* fusion_W  = (const float*)d_in[a++];
  const float* cm_mu_k   = (const float*)d_in[a++];
  const float* cm_mu_r   = (const float*)d_in[a++];
  const float* cm_key_W  = (const float*)d_in[a++];
  const float* cm_rec_W  = (const float*)d_in[a++];
  const float* cm_val_W  = (const float*)d_in[a++];

  // ---------------- workspace carve-up (floats) ----------------
  float* W = (float*)d_ws;  size_t off = 0;
  auto alloc = [&](size_t n){ float* p = W + off; off += n; return p; };
  float* h    = alloc(BNC);
  float* xs   = alloc(BNC);
  float* rb   = alloc(BNC);
  float* kb   = alloc(BNC);
  float* vb   = alloc(BNC);
  float* gate = alloc(BNC);
  float* wb   = alloc(BNC);          // holds c, then decay w in place
  float* fwd  = alloc(BNC);
  float* bwd  = alloc(BNC);
  float* gnb  = alloc(BNC);
  float* ifm  = alloc(BNC);
  float* fb   = alloc(BNC);
  float* t1   = alloc(BNC);
  float* agg  = alloc(BNC);
  float* loc  = alloc(BNC);
  float* xb   = alloc(BNC);
  float* hid  = alloc((size_t)BN*HIDc);
  float* dlt  = alloc((size_t)BN*4);
  float* cf   = alloc(6*Cc);
  int*   knn  = (int*)alloc(BNK);
  float* Eb   = alloc((size_t)BNK*EKP);
  float* E2   = alloc((size_t)BNK*Cc);
  float* E3   = Eb;                  // reuse: E no longer needed once E2 built

  // ---------------- pack all weights to bf16 WMMA fragments ----------------
  auto packB = [&](const float* src, int K, int N)->const __bf16* {
    const int KT = (K+31)/32, NTp = ((N+63)/64)*4;
    const size_t nbf = (size_t)KT*NTp*512;
    __bf16* dst = (__bf16*)(W + off); off += nbf/2;
    k_packB<<<KT*NTp, 256, 0, stream>>>(src, dst, K, N, NTp);
    return dst;
  };
  const __bf16* pWr   = packB(Wr, Cc, Cc);
  const __bf16* pWk   = packB(Wk, Cc, Cc);
  const __bf16* pWv   = packB(Wv, Cc, Cc);
  const __bf16* pWg   = packB(Wg, Cc, Cc);
  const __bf16* pWw   = packB(Ww, Cc, Cc);
  const __bf16* pWo   = packB(Wo, Cc, Cc);
  const __bf16* pOff1 = packB(off_W1, Cc, Cc);
  const __bf16* pOff2 = packB(off_W2, Cc, 3);
  const __bf16 *pE1[ITER], *pE2[ITER], *pU1a[ITER], *pU1b[ITER], *pU2[ITER];
  for (int t = 0; t < ITER; ++t) {
    pE1[t]  = packB(edge_W1 + (size_t)t*EK*Cc, EK, Cc);
    pE2[t]  = packB(edge_W2 + (size_t)t*Cc*Cc, Cc, Cc);
    pU1a[t] = packB(upd_W1 + (size_t)t*2*Cc*Cc,                 Cc, Cc);
    pU1b[t] = packB(upd_W1 + (size_t)t*2*Cc*Cc + (size_t)Cc*Cc, Cc, Cc);
    pU2[t]  = packB(upd_W2 + (size_t)t*Cc*Cc, Cc, Cc);
  }
  const __bf16* pP1   = packB(proj_W1, Cc, Cc);
  const __bf16* pP2   = packB(proj_W2, Cc, Cc);
  const __bf16* pFa   = packB(fusion_W,                 Cc, Cc);
  const __bf16* pFb   = packB(fusion_W + (size_t)Cc*Cc, Cc, Cc);
  const __bf16* pCk   = packB(cm_key_W, Cc, HIDc);
  const __bf16* pCr   = packB(cm_rec_W, Cc, Cc);
  const __bf16* pCv   = packB(cm_val_W, HIDc, Cc);

  auto gemm = [&](const float* A_, const float* XS_, const float* cf_,
                  const __bf16* Bp_, const float* bi_, float* D_,
                  int M, int N, int K, int lda, int ldd, int acc, int epi){
    const int KT = (K+31)/32, NTp = ((N+63)/64)*4;
    dim3 g((N+63)/64, M/128);
    if (XS_) k_gemm_t<1><<<g, 256, 0, stream>>>(A_, XS_, cf_, Bp_, bi_, D_,
                                                N, KT, NTp, lda, ldd, acc, epi);
    else     k_gemm_t<0><<<g, 256, 0, stream>>>(A_, nullptr, nullptr, Bp_, bi_, D_,
                                                N, KT, NTp, lda, ldd, acc, epi);
  };
  const int EB = (int)((BNC + 255)/256);

  // ---------- h = LN1(features); x_star; BQE coefs ----------
  k_layernorm<<<BN/8, 256, 0, stream>>>(features, norm1_g, norm1_b, h, BN);
  k_xstar<<<EB, 256, 0, stream>>>(h, xs);
  k_coef<<<1, Cc, 0, stream>>>(mu_r, mu_k, mu_v, mu_w, cm_mu_k, cm_mu_r, cf);

  // ---------- attention projections (BQE fused into A-load) ----------
  gemm(h, xs, cf+0*Cc, pWr, nullptr, rb,   BN, Cc, Cc, Cc, Cc, 0, 0);
  gemm(h, xs, cf+1*Cc, pWk, nullptr, kb,   BN, Cc, Cc, Cc, Cc, 0, 0);
  gemm(h, xs, cf+2*Cc, pWv, nullptr, vb,   BN, Cc, Cc, Cc, Cc, 0, 0);
  gemm(h, nullptr, nullptr, pWg, nullptr, gate, BN, Cc, Cc, Cc, Cc, 0, 2);
  gemm(h, xs, cf+3*Cc, pWw, nullptr, wb,   BN, Cc, Cc, Cc, Cc, 0, 0);
  k_decay<<<EB, 256, 0, stream>>>(wb, decay_A, decay_B, decay_bse);

  // ---------- bidirectional WKV scan + GN*gate + Wo ----------
  k_wkv<<<dim3(Bb,Hh), 64, 0, stream>>>(rb, kb, vb, wb, uvec, fwd, 0);
  k_wkv<<<dim3(Bb,Hh), 64, 0, stream>>>(rb, kb, vb, wb, uvec, bwd, 1);
  k_gn <<<dim3(Bb,Hh), 256, 0, stream>>>(fwd, bwd, gate, gn_g, gn_b, gnb);
  gemm(gnb, nullptr, nullptr, pWo, nullptr, ifm, BN, Cc, Cc, Cc, Cc, 0, 0);

  // ---------- LGM branch ----------
  k_knn<<<dim3(Nn/128, Bb), 128, 0, stream>>>(xyz, knn);
  (void)hipMemcpyAsync(fb, h, BNC*sizeof(float), hipMemcpyDeviceToDevice, stream);
  for (int t = 0; t < ITER; ++t) {
    gemm(fb, nullptr, nullptr, pOff1, off_b1, t1,  BN, Cc, Cc, Cc, Cc, 0, 1);
    gemm(t1, nullptr, nullptr, pOff2, off_b2, dlt, BN, 3,  Cc, Cc, 4,  0, 0);
    k_edge<<<BN, 256, 0, stream>>>(xyz, dlt, fb, knn, Eb);
    gemm(Eb, nullptr, nullptr, pE1[t], edge_b1 + t*Cc, E2, BNK, Cc, EK, EKP, Cc, 0, 1);
    gemm(E2, nullptr, nullptr, pE2[t], edge_b2 + t*Cc, E3, BNK, Cc, Cc, Cc, Cc, 0, 0);
    k_rowmax<<<EB, 256, 0, stream>>>(E3, agg);
    gemm(agg, nullptr, nullptr, pU1a[t], upd_b1 + t*Cc, t1, BN, Cc, Cc, Cc, Cc, 0, 0);
    gemm(fb,  nullptr, nullptr, pU1b[t], nullptr,       t1, BN, Cc, Cc, Cc, Cc, 1, 1);
    gemm(t1,  nullptr, nullptr, pU2[t],  upd_b2 + t*Cc, fb, BN, Cc, Cc, Cc, Cc, 1, 0);
  }
  gemm(fb, nullptr, nullptr, pP1, proj_b1, t1,  BN, Cc, Cc, Cc, Cc, 0, 4);
  gemm(t1, nullptr, nullptr, pP2, proj_b2, loc, BN, Cc, Cc, Cc, Cc, 0, 0);

  // ---------- fusion: x = features + [ifm|loc] @ fusion_W ----------
  gemm(ifm, nullptr, nullptr, pFa, nullptr, xb, BN, Cc, Cc, Cc, Cc, 0, 0);
  gemm(loc, nullptr, nullptr, pFb, nullptr, xb, BN, Cc, Cc, Cc, Cc, 1, 0);
  k_add2<<<EB, 256, 0, stream>>>(xb, features);

  // ---------- channel mixing ----------
  k_layernorm<<<BN/8, 256, 0, stream>>>(xb, norm2_g, norm2_b, t1, BN);
  k_xstar<<<EB, 256, 0, stream>>>(t1, xs);
  gemm(t1,  xs, cf+4*Cc, pCk, nullptr, hid, BN, HIDc, Cc,  Cc,   HIDc, 0, 3);
  gemm(hid, nullptr, nullptr, pCv, nullptr, vb, BN, Cc, HIDc, HIDc, Cc, 0, 0);
  gemm(t1,  xs, cf+5*Cc, pCr, nullptr, rb,  BN, Cc,  Cc,  Cc,   Cc,  0, 2);
  k_final<<<EB, 256, 0, stream>>>(xb, rb, vb, (float*)d_out);
}
